// GCNClassifier_52355651338769
// MI455X (gfx1250) — matrix operand order repaired and checked by reading, not compiled, not run on hardware
//
#include <hip/hip_runtime.h>
#include <hip/hip_fp16.h>
#include <math.h>

// ---------------- problem constants (match reference) ----------------
#define NNODES 50000
#define NEDGES 800000
#define NGRAPH 256
#define FIN    128
#define HDIM   64
#define NCLS   10

// LDS row stride in halves (K_max=128 + 8 pad -> kills bank conflicts, keeps 16B align)
#define LDA 136

typedef _Float16 v16h __attribute__((ext_vector_type(16)));
typedef _Float16 h8   __attribute__((ext_vector_type(8)));
typedef float    v8f  __attribute__((ext_vector_type(8)));

// =====================================================================
// Degree / symmetric normalization:  dinv[v] = rsqrt(1 + #edges with col==v)
// =====================================================================
__global__ void k_init_deg(float* __restrict__ deg) {
    int v = blockIdx.x * blockDim.x + threadIdx.x;
    if (v < NNODES) deg[v] = 1.0f;                 // self loop
}

__global__ void k_deg_scatter(const int* __restrict__ ei, float* __restrict__ deg) {
    int e = blockIdx.x * blockDim.x + threadIdx.x;
    if (e < NEDGES) {
        int c = ei[NEDGES + e];                    // target / aggregation index
        atomicAdd(&deg[c], 1.0f);
    }
}

__global__ void k_finish_dinv(float* __restrict__ deg) {
    int v = blockIdx.x * blockDim.x + threadIdx.x;
    if (v < NNODES) deg[v] = rsqrtf(deg[v]);       // deg >= 1 always
}

// =====================================================================
// WMMA GEMM:  out[N x 64] = A[N x K] * W[K x 64]   (f16 inputs, f32 accum)
// Block = 256 threads = 8 wave32s; block tile = 128 rows x 64 cols.
// Each wave: 16 rows x 64 cols = 4x v_wmma_f32_16x16x32_f16 per K-step,
// A fragment reused across the 4 N-tiles.
// =====================================================================
__global__ __launch_bounds__(256)
void k_gemm_wmma(const float* __restrict__ A, const float* __restrict__ W,
                 float* __restrict__ out, int N, int K) {
    __shared__ alignas(16) _Float16 sA[128 * LDA];   // activations, row-major in K
    __shared__ alignas(16) _Float16 sB[ 64 * LDA];   // weights transposed: [n][k]

    const int tid     = threadIdx.x;
    const int rowBase = blockIdx.x * 128;

    // stage activation tile (f32 -> f16), zero-fill rows past N
    for (int idx = tid; idx < 128 * K; idx += 256) {
        int r = idx / K, k = idx - r * K;
        int grow = rowBase + r;
        float v = (grow < N) ? A[(long)grow * K + k] : 0.0f;
        sA[r * LDA + k] = (_Float16)v;
    }
    // stage W transposed (f32 -> f16)
    for (int idx = tid; idx < K * 64; idx += 256) {
        int k = idx >> 6, n = idx & 63;
        sB[n * LDA + k] = (_Float16)W[k * 64 + n];
    }
    __syncthreads();

    const int lane = tid & 31;          // wave32
    const int wave = tid >> 5;
    const int m    = lane & 15;
    const int hi   = lane >> 4;
    const int mrow = wave * 16 + m;

    v8f c[4] = {v8f{}, v8f{}, v8f{}, v8f{}};

    for (int kk = 0; kk < K; kk += 32) {
        // A fragment (ISA 7.12.2, 16-bit A 16x32): lanes 0-15 K={0..7,16..23},
        // lanes 16-31 K={8..15,24..31} -> two 16B ds_load_b128 per lane.
        v16h a;
        {
            h8 alo = *(const h8*)&sA[mrow * LDA + kk + hi * 8];
            h8 ahi = *(const h8*)&sA[mrow * LDA + kk + 16 + hi * 8];
            #pragma unroll
            for (int j = 0; j < 8; ++j) { a[j] = alo[j]; a[8 + j] = ahi[j]; }
        }
        #pragma unroll
        for (int t = 0; t < 4; ++t) {
            // B fragment (32x16): lane column n=m, K = {hi*8..hi*8+7, 16+hi*8..16+hi*8+7}
            // (same order as A) in transposed LDS -> two ds_load_b128.
            const _Float16* bp = &sB[(t * 16 + m) * LDA + kk + hi * 8];
            h8 blo = *(const h8*)bp;
            h8 bhi = *(const h8*)(bp + 16);
            v16h b;
            #pragma unroll
            for (int j = 0; j < 8; ++j) { b[j] = blo[j]; b[8 + j] = bhi[j]; }
            c[t] = __builtin_amdgcn_wmma_f32_16x16x32_f16(
                       false, a, false, b, (short)0, c[t], false, false);
        }
    }

    // D layout: VGPR r -> M = r + 8*hi, N = m
    #pragma unroll
    for (int r = 0; r < 8; ++r) {
        int grow = rowBase + wave * 16 + hi * 8 + r;
        if (grow < N) {
            float* o = out + (long)grow * 64 + m;
            o[0]  = c[0][r];
            o[16] = c[1][r];
            o[32] = c[2][r];
            o[48] = c[3][r];
        }
    }
}

// =====================================================================
// agg[v] = t[v]*dinv[v]^2 + b   (self-loop contribution + bias, no atomics)
// =====================================================================
__global__ void k_init_agg(const float* __restrict__ t, const float* __restrict__ dinv,
                           const float* __restrict__ bias, float* __restrict__ agg) {
    int idx = blockIdx.x * blockDim.x + threadIdx.x;
    if (idx < NNODES * HDIM) {
        int v = idx >> 6, f = idx & 63;
        float d = dinv[v];
        agg[idx] = t[idx] * d * d + bias[f];
    }
}

// thread = (edge, feature): coalesced 256B gather of t[row], coalesced f32 atomics to agg[col]
__global__ void k_edge_scatter(const int* __restrict__ ei, const float* __restrict__ t,
                               const float* __restrict__ dinv, float* __restrict__ agg) {
    long gid = (long)blockIdx.x * blockDim.x + threadIdx.x;
    if (gid < (long)NEDGES * HDIM) {
        int e = (int)(gid >> 6), f = (int)(gid & 63);
        int r = ei[e];
        int cidx = ei[NEDGES + e];
        float w = dinv[r] * dinv[cidx];
        atomicAdd(&agg[cidx * HDIM + f], t[r * HDIM + f] * w);
    }
}

// =====================================================================
// L2-normalize row then ReLU. One wave32 per node; lane owns features f and f+32.
// =====================================================================
__global__ __launch_bounds__(256)
void k_l2relu(const float* __restrict__ in, float* __restrict__ out, int N) {
    int wave = threadIdx.x >> 5;
    int lane = threadIdx.x & 31;
    int v = blockIdx.x * 8 + wave;
    if (v < N) {
        float a = in[v * HDIM + lane];
        float b = in[v * HDIM + lane + 32];
        float ss = a * a + b * b;
        #pragma unroll
        for (int msk = 16; msk >= 1; msk >>= 1)
            ss += __shfl_xor(ss, msk, 32);
        float scale = 1.0f / fmaxf(sqrtf(ss), 1e-12f);
        out[v * HDIM + lane]      = fmaxf(a * scale, 0.0f);
        out[v * HDIM + lane + 32] = fmaxf(b * scale, 0.0f);
    }
}

// =====================================================================
// Global mean pool + classifier + softmax
// =====================================================================
__global__ void k_zero_pool(float* __restrict__ sums, float* __restrict__ cnts) {
    int idx = blockIdx.x * blockDim.x + threadIdx.x;
    if (idx < NGRAPH * HDIM) sums[idx] = 0.0f;
    if (idx < NGRAPH) cnts[idx] = 0.0f;
}

__global__ void k_pool_scatter(const float* __restrict__ h, const int* __restrict__ batch,
                               float* __restrict__ sums, float* __restrict__ cnts) {
    int idx = blockIdx.x * blockDim.x + threadIdx.x;
    if (idx < NNODES * HDIM) {
        int v = idx >> 6, f = idx & 63;
        int g = batch[v];
        atomicAdd(&sums[g * HDIM + f], h[idx]);
        if (f == 0) atomicAdd(&cnts[g], 1.0f);
    }
}

__global__ __launch_bounds__(64)
void k_graph_finalize(const float* __restrict__ sums, const float* __restrict__ cnts,
                      const float* __restrict__ Wm, const float* __restrict__ bm,
                      float* __restrict__ graph_emb, float* __restrict__ logits,
                      float* __restrict__ probs) {
    __shared__ float sge[HDIM];
    __shared__ float slg[NCLS];
    int g = blockIdx.x;
    int f = threadIdx.x;

    float denom = fmaxf(cnts[g], 1.0f);
    float ge = sums[g * HDIM + f] / denom;
    graph_emb[g * HDIM + f] = ge;
    sge[f] = ge;
    __syncthreads();

    if (f < NCLS) {
        float acc = bm[f];
        #pragma unroll
        for (int k = 0; k < HDIM; ++k) acc += sge[k] * Wm[k * NCLS + f];
        logits[g * NCLS + f] = acc;
        slg[f] = acc;
    }
    __syncthreads();

    if (f < NCLS) {
        float mx = slg[0];
        for (int k = 1; k < NCLS; ++k) mx = fmaxf(mx, slg[k]);
        float se = 0.0f;
        for (int k = 0; k < NCLS; ++k) se += expf(slg[k] - mx);
        probs[g * NCLS + f] = expf(slg[f] - mx) / se;
    }
}

// =====================================================================
// Launch sequence
// =====================================================================
extern "C" void kernel_launch(void* const* d_in, const int* in_sizes, int n_in,
                              void* d_out, int out_size, void* d_ws, size_t ws_size,
                              hipStream_t stream) {
    const float* x     = (const float*)d_in[0];
    const int*   ei    = (const int*)  d_in[1];   // [2, NEDGES] flat
    const int*   batch = (const int*)  d_in[2];
    const float* W1 = (const float*)d_in[3];
    const float* b1 = (const float*)d_in[4];
    const float* W2 = (const float*)d_in[5];
    const float* b2 = (const float*)d_in[6];
    const float* W3 = (const float*)d_in[7];
    const float* b3 = (const float*)d_in[8];
    const float* Wm = (const float*)d_in[9];
    const float* bm = (const float*)d_in[10];

    // output layout: logits | probs | node_emb | graph_emb
    float* out       = (float*)d_out;
    float* logits    = out;
    float* probs     = out + NGRAPH * NCLS;
    float* node_emb  = probs + NGRAPH * NCLS;
    float* graph_emb = node_emb + (size_t)NNODES * HDIM;

    // workspace layout (floats)
    float* ws   = (float*)d_ws;
    float* dinv = ws;                              // NNODES (padded to 50048)
    float* tbuf = ws + 50048;                      // NNODES*64  (GEMM result)
    float* agg  = tbuf + (size_t)NNODES * HDIM;    // NNODES*64  (aggregation)
    float* hbuf = agg  + (size_t)NNODES * HDIM;    // NNODES*64  (layer activations)
    float* sums = hbuf + (size_t)NNODES * HDIM;    // 256*64
    float* cnts = sums + NGRAPH * HDIM;            // 256

    const int T = 256;

    // symmetric normalization coefficients
    k_init_deg   <<<(NNODES + T - 1) / T, T, 0, stream>>>(dinv);
    k_deg_scatter<<<(NEDGES + T - 1) / T, T, 0, stream>>>(ei, dinv);
    k_finish_dinv<<<(NNODES + T - 1) / T, T, 0, stream>>>(dinv);

    const float* Ws[3] = {W1, W2, W3};
    const float* bs[3] = {b1, b2, b3};
    const float* hin = x;
    for (int l = 0; l < 3; ++l) {
        int K = (l == 0) ? FIN : HDIM;
        k_gemm_wmma<<<(NNODES + 127) / 128, T, 0, stream>>>(hin, Ws[l], tbuf, NNODES, K);
        k_init_agg <<<(NNODES * HDIM + T - 1) / T, T, 0, stream>>>(tbuf, dinv, bs[l], agg);
        k_edge_scatter<<<(int)(((long)NEDGES * HDIM + T - 1) / T), T, 0, stream>>>(ei, tbuf, dinv, agg);
        float* hout = (l == 2) ? node_emb : hbuf;
        k_l2relu<<<(NNODES + 7) / 8, T, 0, stream>>>(agg, hout, NNODES);
        hin = hout;
    }

    k_zero_pool   <<<(NGRAPH * HDIM + T - 1) / T, T, 0, stream>>>(sums, cnts);
    k_pool_scatter<<<(NNODES * HDIM + T - 1) / T, T, 0, stream>>>(node_emb, batch, sums, cnts);
    k_graph_finalize<<<NGRAPH, 64, 0, stream>>>(sums, cnts, Wm, bm, graph_emb, logits, probs);
}
